// TTTWithAdaptiveNorm_39444979646896
// MI455X (gfx1250) — matrix operand order repaired and checked by reading, not compiled, not run on hardware
//
#include <hip/hip_runtime.h>
#include <hip/hip_bf16.h>
#include <cstdint>

typedef _Float16 v8h  __attribute__((ext_vector_type(8)));
typedef _Float16 v16h __attribute__((ext_vector_type(16)));
typedef float    v8f  __attribute__((ext_vector_type(8)));

// ---------------------------------------------------------------------------
// WMMA helpers (V_WMMA_F32_16X16X32_F16, D = A(16x32,f16) * B(32x16,f16) + C)
// A layout used here (contiguous k, not the ISA table order): lane<16 -> M=lane, halves = K {k0..k0+15}
//           lane>=16 -> M=lane-16, halves = K {k0+16..k0+31}
// B layout used here (same k order as A): lane<16 -> N=lane,  halves = K {k0..k0+15}
//           lane>=16 -> N=lane-16, halves = K {k0+16..k0+31}
// C/D layout: VGPR r, lane l -> row M = r + 8*(l>=16), col N = l%16
// ---------------------------------------------------------------------------
__device__ inline v8f wmma16(v16h a, v16h b, v8f c) {
  return __builtin_amdgcn_wmma_f32_16x16x32_f16(false, a, false, b, (short)0, c, false, false);
}

__device__ inline v16h load_frag_A(const _Float16* base, int lda, int m, int half, int k0) {
  const _Float16* p = base + (size_t)m * lda + k0 + half * 16;
  v8h lo = *(const v8h*)p;
  v8h hi = *(const v8h*)(p + 8);
  v16h r;
#pragma unroll
  for (int i = 0; i < 8; ++i) { r[i] = lo[i]; r[i + 8] = hi[i]; }
  return r;
}

__device__ inline v16h load_frag_B(const _Float16* base, int ldb, int n, int half, int k0) {
  const _Float16* p = base + (size_t)n * ldb + k0 + half * 16;
  v8h lo = *(const v8h*)p;
  v8h hi = *(const v8h*)(p + 8);
  v16h r;
#pragma unroll
  for (int i = 0; i < 8; ++i) { r[i] = lo[i]; r[i + 8] = hi[i]; }
  return r;
}

// CDNA5 async global->LDS copy (ASYNCcnt-tracked), GVS addressing:
// LDS[vdst] = MEM[saddr + voff], 16 bytes per active lane.
__device__ inline void async_copy_b128(uint32_t lds_byte_addr, uint64_t gbase, uint32_t gbyte_off) {
  asm volatile("global_load_async_to_lds_b128 %0, %1, %2"
               :: "v"(lds_byte_addr), "v"(gbyte_off), "s"(gbase)
               : "memory");
}
#define WAIT_ASYNC(n) asm volatile("s_wait_asynccnt " #n ::: "memory")

// generic flat pointer to LDS: low 32 bits are the LDS byte offset
__device__ inline uint32_t lds_addr32(const void* p) { return (uint32_t)(uintptr_t)p; }

// sum across the 16-lane half-group (lanes 0-15 and 16-31 reduce separately)
__device__ inline float redsum16(float x) {
#pragma unroll
  for (int off = 1; off < 16; off <<= 1) x += __shfl_xor(x, off, 32);
  return x;
}

// ---------------------------------------------------------------------------
// elementwise converters
// ---------------------------------------------------------------------------
__global__ void cvt_f16_kernel(const float* __restrict__ in, _Float16* __restrict__ out, int n) {
  int i = blockIdx.x * blockDim.x + threadIdx.x;
  if (i < n) out[i] = (_Float16)in[i];
}

// W1 [H,D,D] (d1,f) -> W1T16 [H, f, d1] f16  (B-operand wants column-contiguous)
__global__ void cvt_w1t_kernel(const float* __restrict__ W1, _Float16* __restrict__ W1T) {
  int i = blockIdx.x * blockDim.x + threadIdx.x;
  if (i < 16 * 64 * 64) {
    int h = i >> 12, d1 = (i >> 6) & 63, f = i & 63;
    W1T[(h << 12) + (f << 6) + d1] = (_Float16)W1[i];
  }
}

// XQf/XKf fp32 [B,H,L,D] -> XQ16/XK16 f16 (same layout), XK16T f16 [B,H,D,L]
__global__ void cvt_qk_kernel(const float* __restrict__ XQf, const float* __restrict__ XKf,
                              _Float16* __restrict__ XQ16, _Float16* __restrict__ XK16,
                              _Float16* __restrict__ XK16T) {
  size_t i = (size_t)blockIdx.x * blockDim.x + threadIdx.x;
  if (i < (size_t)32 * 2048 * 64) {
    XQ16[i] = (_Float16)XQf[i];
    float k = XKf[i];
    XK16[i] = (_Float16)k;
    size_t bh = i >> 17, l = (i >> 6) & 2047, d = i & 63;
    XK16T[(bh << 17) + (d << 11) + l] = (_Float16)k;
  }
}

// ---------------------------------------------------------------------------
// eta[b,h,l] = sigmoid(hidden[b,l,:].lr_weight[h,:,0] + lr_bias[h]) / 64
// ---------------------------------------------------------------------------
__global__ void eta_kernel(const float* __restrict__ hidden, const float* __restrict__ lrw,
                           const float* __restrict__ lrb, float* __restrict__ eta) {
  int gw = blockIdx.x * (blockDim.x >> 5) + (threadIdx.x >> 5);  // [0, 4096*16)
  int lane = threadIdx.x & 31;
  int h = gw & 15, t = gw >> 4;  // t = b*2048 + l
  const float* x = hidden + (size_t)t * 1024;
  const float* w = lrw + (size_t)h * 1024;
  float s = 0.f;
  for (int c = lane; c < 1024; c += 32) s += x[c] * w[c];
#pragma unroll
  for (int off = 1; off < 32; off <<= 1) s += __shfl_xor(s, off, 32);
  if (lane == 0) {
    float sig = 1.f / (1.f + expf(-(s + lrb[h])));
    int b = t >> 11, l = t & 2047;
    eta[((size_t)(b * 16 + h)) * 2048 + l] = sig * (1.f / 64.f);
  }
}

// ---------------------------------------------------------------------------
// Projections: Y = X @ W^T, X:[4096,1024] f16, W:[1024,1024] f16
// Block = 8 waves = 128 rows x 64 cols (one head). W panel (64x32 per K-step)
// is double-buffered through LDS via global_load_async_to_lds_b128 and shared
// by all 8 waves; A fragments are register-pipelined global b128 loads.
// ---------------------------------------------------------------------------
__global__ void gemm_proj_kernel(const _Float16* __restrict__ X,
                                 const _Float16* __restrict__ Wq16,
                                 const _Float16* __restrict__ Wk16,
                                 const _Float16* __restrict__ Wv16,
                                 float* __restrict__ XQf, float* __restrict__ XKf,
                                 float* __restrict__ XVf) {
  __shared__ __align__(16) _Float16 shB[2][64 * 32];  // 2 x 4KB
  const int tid = threadIdx.x;
  const int wave = tid >> 5, lane = tid & 31;
  const int half = lane >> 4, l15 = lane & 15;
  const _Float16* W;
  float* Y;
  if (blockIdx.z == 0)      { W = Wq16; Y = XQf; }
  else if (blockIdx.z == 1) { W = Wk16; Y = XKf; }
  else                      { W = Wv16; Y = XVf; }
  const int mBase = blockIdx.x * 128 + wave * 16;
  const int h = blockIdx.y;
  const int nBase = h * 64;

  // async B staging: thread tid copies 16B: row bn (W col nBase+bn), k chunk bk
  const int bn = tid >> 2;            // 0..63
  const int bk = (tid & 3) << 3;      // 0,8,16,24
  const uint32_t gOffBase = (((uint32_t)(nBase + bn) << 10) + (uint32_t)bk) << 1;  // bytes
  const uint64_t gW = (uint64_t)(uintptr_t)W;
  const uint32_t ldsDst[2] = { lds_addr32(&shB[0][(bn << 5) + bk]),
                               lds_addr32(&shB[1][(bn << 5) + bk]) };
  // prologue: stages 0 and 1 in flight
  async_copy_b128(ldsDst[0], gW, gOffBase);
  async_copy_b128(ldsDst[1], gW, gOffBase + 64);

  const _Float16* Arow = X + (size_t)(mBase + l15) * 1024 + half * 16;
  auto loadA = [&](int k0) -> v16h {
    v8h lo = *(const v8h*)(Arow + k0);
    v8h hi = *(const v8h*)(Arow + k0 + 8);
    v16h r;
#pragma unroll
    for (int i = 0; i < 8; ++i) { r[i] = lo[i]; r[i + 8] = hi[i]; }
    return r;
  };

  v8f acc[4] = {};
  v16h aCur = loadA(0);
  for (int s = 0; s < 31; ++s) {
    v16h aNext = loadA(32 * (s + 1));      // register-pipelined A prefetch
    WAIT_ASYNC(0x1);                       // stage s resident, s+1 may be in flight
    __syncthreads();
    const _Float16* bbuf = shB[s & 1];
#pragma unroll
    for (int j = 0; j < 4; ++j) {
      const _Float16* p = bbuf + ((j * 16 + l15) << 5) + (half << 4);
      v8h lo = *(const v8h*)p;
      v8h hi = *(const v8h*)(p + 8);
      v16h b;
#pragma unroll
      for (int i = 0; i < 8; ++i) { b[i] = lo[i]; b[i + 8] = hi[i]; }
      acc[j] = wmma16(aCur, b, acc[j]);
    }
    __syncthreads();                       // everyone done reading buf s&1
    if (s < 30) async_copy_b128(ldsDst[s & 1], gW, gOffBase + (uint32_t)(32 * (s + 2)) * 2u);
    aCur = aNext;
  }
  // peeled last stage (s = 31, buffer 1)
  WAIT_ASYNC(0x0);
  __syncthreads();
  {
    const _Float16* bbuf = shB[1];
#pragma unroll
    for (int j = 0; j < 4; ++j) {
      const _Float16* p = bbuf + ((j * 16 + l15) << 5) + (half << 4);
      v8h lo = *(const v8h*)p;
      v8h hi = *(const v8h*)(p + 8);
      v16h b;
#pragma unroll
      for (int i = 0; i < 8; ++i) { b[i] = lo[i]; b[i + 8] = hi[i]; }
      acc[j] = wmma16(aCur, b, acc[j]);
    }
  }
#pragma unroll
  for (int j = 0; j < 4; ++j)
#pragma unroll
    for (int r = 0; r < 8; ++r) {
      int row = mBase + r + half * 8;  // t in [0,4096)
      int b = row >> 11, l = row & 2047;
      int d = j * 16 + l15;
      Y[(((size_t)(b * 16 + h)) * 2048 + l) * 64 + d] = acc[j][r];
    }
}

// ---------------------------------------------------------------------------
// Z1 = XK@W1 + b1 ; LN fused L2 bwd vs (XV-XK) ; eg = eta*grad -> eg16T [bh,d,l]
// W1T tile (64x64, 8KB) async-staged to LDS once, shared by all 8 waves.
// ---------------------------------------------------------------------------
__global__ void z1_eg_kernel(const _Float16* __restrict__ XK16, const _Float16* __restrict__ W1T16,
                             const float* __restrict__ XKf, const float* __restrict__ XVf,
                             const float* __restrict__ b1, const float* __restrict__ gamma,
                             const float* __restrict__ beta, const float* __restrict__ eta,
                             _Float16* __restrict__ eg16T) {
  __shared__ __align__(16) _Float16 shW[64 * 64];  // 8KB
  const int tid = threadIdx.x;
  const int wave = tid >> 5, lane = tid & 31;
  const int half = lane >> 4, l15 = lane & 15;
  const int bh = blockIdx.y, h = bh & 15;
  const int lBase = blockIdx.x * 128 + wave * 16;
  const _Float16* A = XK16 + (size_t)bh * 2048 * 64;
  const uint64_t gB = (uint64_t)(uintptr_t)(W1T16 + (size_t)h * 64 * 64);

  {
    uint32_t off = (uint32_t)tid * 32u;  // bytes: 16 halves per thread
    uint32_t dst = lds_addr32(shW) + off;
    async_copy_b128(dst, gB, off);
    async_copy_b128(dst + 16, gB, off + 16);
  }
  WAIT_ASYNC(0x0);
  __syncthreads();

  v8f acc[4] = {};
#pragma unroll
  for (int k0 = 0; k0 < 64; k0 += 32) {
    v16h a = load_frag_A(A + (size_t)lBase * 64, 64, l15, half, k0);
#pragma unroll
    for (int j = 0; j < 4; ++j) {
      v16h b = load_frag_B(shW + (size_t)(j * 16) * 64, 64, l15, half, k0);
      acc[j] = wmma16(a, b, acc[j]);
    }
  }
  float g[4], bt[4], b1v[4];
#pragma unroll
  for (int t = 0; t < 4; ++t) {
    int d = t * 16 + l15;
    g[t] = gamma[h * 64 + d]; bt[t] = beta[h * 64 + d]; b1v[t] = b1[h * 64 + d];
  }
  const float* xkb = XKf + (size_t)bh * 2048 * 64;
  const float* xvb = XVf + (size_t)bh * 2048 * 64;
#pragma unroll
  for (int r = 0; r < 8; ++r) {
    int row = lBase + r + half * 8;
    float z[4], tgt[4];
#pragma unroll
    for (int t = 0; t < 4; ++t) {
      int d = t * 16 + l15;
      z[t] = acc[t][r] + b1v[t];
      tgt[t] = xvb[(size_t)row * 64 + d] - xkb[(size_t)row * 64 + d];
    }
    float mu = redsum16(z[0] + z[1] + z[2] + z[3]) * (1.f / 64.f);
    float xh[4], vs = 0.f;
#pragma unroll
    for (int t = 0; t < 4; ++t) { float c = z[t] - mu; xh[t] = c; vs += c * c; }
    float var = redsum16(vs) * (1.f / 64.f);
    float rstd = 1.f / sqrtf(var + 1e-6f);
    float gxh[4], m1 = 0.f, m2 = 0.f;
#pragma unroll
    for (int t = 0; t < 4; ++t) {
      xh[t] *= rstd;
      gxh[t] = (g[t] * xh[t] + bt[t] - tgt[t]) * g[t];
      m1 += gxh[t]; m2 += gxh[t] * xh[t];
    }
    m1 = redsum16(m1) * (1.f / 64.f);
    m2 = redsum16(m2) * (1.f / 64.f);
    float etav = eta[(size_t)bh * 2048 + row];
#pragma unroll
    for (int t = 0; t < 4; ++t) {
      float e = etav * ((gxh[t] - m1 - xh[t] * m2) * rstd);
      int d = t * 16 + l15;
      eg16T[((size_t)bh * 64 + d) * 2048 + row] = (_Float16)e;
    }
  }
}

// b1sum[bh*64+d] = sum_l eg16T[bh,d,l]  (one wave per (bh,d); deterministic)
__global__ void b1sum_kernel(const _Float16* __restrict__ eg16T, float* __restrict__ b1sum) {
  int gw = blockIdx.x * (blockDim.x >> 5) + (threadIdx.x >> 5);  // [0,2048)
  int lane = threadIdx.x & 31;
  const _Float16* p = eg16T + (size_t)gw * 2048;
  float s = 0.f;
  for (int l = lane; l < 2048; l += 32) s += (float)p[l];
#pragma unroll
  for (int off = 1; off < 32; off <<= 1) s += __shfl_xor(s, off, 32);
  if (lane == 0) b1sum[gw] = s;
}

// ---------------------------------------------------------------------------
// Gpart[slice][bh][d1][f] = sum_{l in slice} XK[l,d1]*eg[l,f]
// grid (32, 4), block = 1 wave; wave computes full 64x64 over a K=512 slice
// ---------------------------------------------------------------------------
__global__ void g_kernel(const _Float16* __restrict__ XK16T, const _Float16* __restrict__ eg16T,
                         float* __restrict__ Gpart) {
  const int lane = threadIdx.x & 31, half = lane >> 4, l15 = lane & 15;
  const int bh = blockIdx.x, slice = blockIdx.y;
  const _Float16* A = XK16T + (size_t)bh * 64 * 2048;
  const _Float16* Bm = eg16T + (size_t)bh * 64 * 2048;
  v8f acc[4][4] = {};
  for (int k0 = slice * 512; k0 < slice * 512 + 512; k0 += 32) {
    v16h a[4], b[4];
#pragma unroll
    for (int mi = 0; mi < 4; ++mi) a[mi] = load_frag_A(A + (size_t)(mi * 16) * 2048, 2048, l15, half, k0);
#pragma unroll
    for (int ni = 0; ni < 4; ++ni) b[ni] = load_frag_B(Bm + (size_t)(ni * 16) * 2048, 2048, l15, half, k0);
#pragma unroll
    for (int mi = 0; mi < 4; ++mi)
#pragma unroll
      for (int ni = 0; ni < 4; ++ni) acc[mi][ni] = wmma16(a[mi], b[ni], acc[mi][ni]);
  }
  float* Gp = Gpart + (size_t)slice * 131072 + (size_t)bh * 4096;
#pragma unroll
  for (int mi = 0; mi < 4; ++mi)
#pragma unroll
    for (int ni = 0; ni < 4; ++ni)
#pragma unroll
      for (int r = 0; r < 8; ++r) {
        int d1 = mi * 16 + r + half * 8, f = ni * 16 + l15;
        Gp[d1 * 64 + f] = acc[mi][ni][r];
      }
}

// W1effT[bh,f,d1] = f16(W1[h,d1,f] - sum_slices Gpart); b1bar = b1 - b1sum
__global__ void w1eff_kernel(const float* __restrict__ W1, const float* __restrict__ Gpart,
                             const float* __restrict__ b1, const float* __restrict__ b1sum,
                             _Float16* __restrict__ W1effT, float* __restrict__ b1bar) {
  int i = blockIdx.x * blockDim.x + threadIdx.x;
  if (i < 2 * 16 * 64 * 64) {
    int bh = i >> 12, h = bh & 15, d1 = (i >> 6) & 63, f = i & 63;
    float gsum = Gpart[i] + Gpart[131072 + i] + Gpart[2 * 131072 + i] + Gpart[3 * 131072 + i];
    float v = W1[(h << 12) + (d1 << 6) + f] - gsum;
    W1effT[((size_t)bh << 12) + (f << 6) + d1] = (_Float16)v;
  }
  if (i < 2048) b1bar[i] = b1[((i >> 6) & 15) * 64 + (i & 63)] - b1sum[i];
}

// ---------------------------------------------------------------------------
// out = XQ + LN(XQ@W1_eff + b1_bar); W1_eff tile async-staged to LDS
// ---------------------------------------------------------------------------
__global__ void final_kernel(const _Float16* __restrict__ XQ16, const _Float16* __restrict__ W1effT,
                             const float* __restrict__ XQf, const float* __restrict__ b1bar,
                             const float* __restrict__ gamma, const float* __restrict__ beta,
                             float* __restrict__ out) {
  __shared__ __align__(16) _Float16 shW[64 * 64];  // 8KB
  const int tid = threadIdx.x;
  const int wave = tid >> 5, lane = tid & 31;
  const int half = lane >> 4, l15 = lane & 15;
  const int bh = blockIdx.y, h = bh & 15;
  const int lBase = blockIdx.x * 128 + wave * 16;
  const _Float16* A = XQ16 + (size_t)bh * 2048 * 64;
  const uint64_t gB = (uint64_t)(uintptr_t)(W1effT + (size_t)bh * 64 * 64);

  {
    uint32_t off = (uint32_t)tid * 32u;
    uint32_t dst = lds_addr32(shW) + off;
    async_copy_b128(dst, gB, off);
    async_copy_b128(dst + 16, gB, off + 16);
  }
  WAIT_ASYNC(0x0);
  __syncthreads();

  v8f acc[4] = {};
#pragma unroll
  for (int k0 = 0; k0 < 64; k0 += 32) {
    v16h a = load_frag_A(A + (size_t)lBase * 64, 64, l15, half, k0);
#pragma unroll
    for (int j = 0; j < 4; ++j) {
      v16h b = load_frag_B(shW + (size_t)(j * 16) * 64, 64, l15, half, k0);
      acc[j] = wmma16(a, b, acc[j]);
    }
  }
  float g[4], bt[4], bb[4];
#pragma unroll
  for (int t = 0; t < 4; ++t) {
    int d = t * 16 + l15;
    g[t] = gamma[h * 64 + d]; bt[t] = beta[h * 64 + d]; bb[t] = b1bar[bh * 64 + d];
  }
  const float* xqb = XQf + (size_t)bh * 2048 * 64;
  float* ob = out + (size_t)bh * 2048 * 64;
#pragma unroll
  for (int r = 0; r < 8; ++r) {
    int row = lBase + r + half * 8;
    float z[4];
#pragma unroll
    for (int t = 0; t < 4; ++t) z[t] = acc[t][r] + bb[t];
    float mu = redsum16(z[0] + z[1] + z[2] + z[3]) * (1.f / 64.f);
    float xh[4], vs = 0.f;
#pragma unroll
    for (int t = 0; t < 4; ++t) { float c = z[t] - mu; xh[t] = c; vs += c * c; }
    float var = redsum16(vs) * (1.f / 64.f);
    float rstd = 1.f / sqrtf(var + 1e-6f);
#pragma unroll
    for (int t = 0; t < 4; ++t) {
      int d = t * 16 + l15;
      ob[(size_t)row * 64 + d] = xqb[(size_t)row * 64 + d] + g[t] * (xh[t] * rstd) + bt[t];
    }
  }
}

// ---------------------------------------------------------------------------
extern "C" void kernel_launch(void* const* d_in, const int* in_sizes, int n_in,
                              void* d_out, int out_size, void* d_ws, size_t ws_size,
                              hipStream_t stream) {
  (void)in_sizes; (void)n_in; (void)out_size; (void)ws_size;
  const float* hidden = (const float*)d_in[0];
  const float* Wq     = (const float*)d_in[1];
  const float* Wk     = (const float*)d_in[2];
  const float* Wv     = (const float*)d_in[3];
  const float* W1     = (const float*)d_in[4];
  const float* b1     = (const float*)d_in[5];
  const float* gamma  = (const float*)d_in[6];
  const float* beta   = (const float*)d_in[7];
  const float* lrw    = (const float*)d_in[8];
  const float* lrb    = (const float*)d_in[9];
  float* out = (float*)d_out;

  char* ws = (char*)d_ws;
  size_t off = 0;
  auto alloc = [&](size_t bytes) -> char* {
    char* p = ws + off;
    off += (bytes + 255) & ~(size_t)255;
    return p;
  };
  _Float16* h16    = (_Float16*)alloc((size_t)4096 * 1024 * 2);
  _Float16* wq16   = (_Float16*)alloc((size_t)1024 * 1024 * 2);
  _Float16* wk16   = (_Float16*)alloc((size_t)1024 * 1024 * 2);
  _Float16* wv16   = (_Float16*)alloc((size_t)1024 * 1024 * 2);
  _Float16* w1t16  = (_Float16*)alloc((size_t)65536 * 2);
  float*    XQf    = (float*)alloc((size_t)4194304 * 4);
  float*    XKf    = (float*)alloc((size_t)4194304 * 4);
  float*    XVf    = (float*)alloc((size_t)4194304 * 4);
  _Float16* XQ16   = (_Float16*)alloc((size_t)4194304 * 2);
  _Float16* XK16   = (_Float16*)alloc((size_t)4194304 * 2);
  _Float16* XK16T  = (_Float16*)alloc((size_t)4194304 * 2);
  _Float16* eg16T  = (_Float16*)alloc((size_t)4194304 * 2);
  float*    eta    = (float*)alloc((size_t)65536 * 4);
  float*    Gpart  = (float*)alloc((size_t)4 * 131072 * 4);
  float*    b1sum  = (float*)alloc((size_t)2048 * 4);
  _Float16* w1effT = (_Float16*)alloc((size_t)131072 * 2);
  float*    b1bar  = (float*)alloc((size_t)2048 * 4);

  dim3 b256(256);
  cvt_f16_kernel<<<dim3(16384), b256, 0, stream>>>(hidden, h16, 4194304);
  cvt_f16_kernel<<<dim3(4096), b256, 0, stream>>>(Wq, wq16, 1048576);
  cvt_f16_kernel<<<dim3(4096), b256, 0, stream>>>(Wk, wk16, 1048576);
  cvt_f16_kernel<<<dim3(4096), b256, 0, stream>>>(Wv, wv16, 1048576);
  cvt_w1t_kernel<<<dim3(256), b256, 0, stream>>>(W1, w1t16);
  eta_kernel<<<dim3(8192), b256, 0, stream>>>(hidden, lrw, lrb, eta);
  gemm_proj_kernel<<<dim3(32, 16, 3), b256, 0, stream>>>(h16, wq16, wk16, wv16, XQf, XKf, XVf);
  cvt_qk_kernel<<<dim3(16384), b256, 0, stream>>>(XQf, XKf, XQ16, XK16, XK16T);
  z1_eg_kernel<<<dim3(16, 32), b256, 0, stream>>>(XK16, w1t16, XKf, XVf, b1, gamma, beta, eta, eg16T);
  b1sum_kernel<<<dim3(256), b256, 0, stream>>>(eg16T, b1sum);
  g_kernel<<<dim3(32, 4), dim3(32), 0, stream>>>(XK16T, eg16T, Gpart);
  w1eff_kernel<<<dim3(512), b256, 0, stream>>>(W1, Gpart, b1, b1sum, w1effT, b1bar);
  final_kernel<<<dim3(16, 32), b256, 0, stream>>>(XQ16, w1effT, XQf, b1bar, gamma, beta, out);
}
